// FluidPushModel_68994354643552
// MI455X (gfx1250) — matrix operand order repaired and checked by reading, not compiled, not run on hardware
//
#include <hip/hip_runtime.h>

// ---------------------------------------------------------------------------
// FluidPush for MI455X (gfx1250, wave32).
// 7-tap separable Gaussian blur of the velocity field is expressed as
// 16x16-tile matrix products and executed on the WMMA units:
//   blur-x:  D(16x16) = A_data(16x32) * B_const(32x16)   (one v_wmma per tile)
//   blur-y:  D(16x16) = A_const(16x32) * B_data(32x16)   (one v_wmma per tile)
// where the constant operand is the banded Toeplitz Gaussian (radius 3 fits
// in the K=32 window with an 8-element offset).
//
// Velocity halo staging uses GLOBAL_LOAD_ASYNC_TO_LDS_B128 (ASYNCcnt) when
// the toolchain exposes the builtin; scalar fallback otherwise.
//
// Workspace layout (requires ws_size >= 48 MiB):
//   [0,16M)  : vel ping  (f16, [img][comp][256][256])
//   [16,32M) : vel pong
//   [32,48M) : rho scratch (f32);  d_out is the second rho buffer.
// ---------------------------------------------------------------------------

typedef __attribute__((ext_vector_type(16))) _Float16 v16h;
typedef __attribute__((ext_vector_type(8)))  float    v8f;

#define N_STEPS_C     20
#define DECAY_C       0.95f
#define MEDIA_SHARP_C 5.0f
#define WIDTH_C       5.0f
#define SIGMA_TOOL_C  1.0f
#define IMG_N         256
#define STRIP_R       16

#if defined(__AMDGCN__) &&                                                    \
    __has_builtin(__builtin_amdgcn_global_load_async_to_lds_b128) &&          \
    __has_builtin(__builtin_amdgcn_s_wait_asynccnt)
#define FP_ASYNC_STAGE 1
// Pointer types matching the builtin prototype reported by clang:
//   param 1: AS1 (global) pointer to int __vector_size__(16)
//   param 2: AS3 (LDS)    pointer to int __vector_size__(16)
typedef int fp_v4i __attribute__((vector_size(16)));
typedef __attribute__((address_space(1))) fp_v4i* fp_gp128;
typedef __attribute__((address_space(3))) fp_v4i* fp_lp128;
#else
#define FP_ASYNC_STAGE 0
#endif

static __device__ __forceinline__ float sigmoidf_(float x) {
  return 1.0f / (1.0f + expf(-x));
}
static __device__ __forceinline__ int clampi_(int v, int lo, int hi) {
  return v < lo ? lo : (v > hi ? hi : v);
}

// ---------------------------------------------------------------------------
// Init: swept-segment soft mask -> initial velocity (f16 planes).
// Sim convention: X = occ column (j), Y = occ row (i); comp0 along j, comp1
// along i (this absorbs the occ.T ... .T round trip of the reference).
// ---------------------------------------------------------------------------
__global__ __launch_bounds__(256) void fp_init_vel(
    const float* __restrict__ astart, const float* __restrict__ aend,
    _Float16* __restrict__ vel0) {
  const int img = blockIdx.y;
  const int idx = blockIdx.x * 256 + threadIdx.x;  // 0..65535
  const int i = idx >> 8, j = idx & 255;

  const float s0 = astart[img * 3 + 0], s1 = astart[img * 3 + 1];
  const float d0 = aend[img * 3 + 0] - s0, d1 = aend[img * 3 + 1] - s1;
  const float L2 = d0 * d0 + d1 * d1 + 1e-12f;
  const float X = (float)j, Y = (float)i;
  float t = ((X - s0) * d0 + (Y - s1) * d1) / L2;
  t = fminf(fmaxf(t, 0.0f), 1.0f);
  const float px = s0 + t * d0, py = s1 + t * d1;
  const float dist =
      sqrtf((X - px) * (X - px) + (Y - py) * (Y - py) + 1e-12f);
  const float mask = sigmoidf_((WIDTH_C - dist) / SIGMA_TOOL_C);
  const float invN = 1.0f / (float)N_STEPS_C;

  const size_t base = (size_t)img * 2u * 65536u;
  vel0[base + idx]          = (_Float16)(mask * d0 * invN);
  vel0[base + 65536u + idx] = (_Float16)(mask * d1 * invN);
}

// ---------------------------------------------------------------------------
// One simulation step: blur(vel)*decay*gate(rho) then advect rho.
// Grid: (16 strips, 64 images), 256 threads = 8 wave32.
// ---------------------------------------------------------------------------
__global__ __launch_bounds__(256) void fp_step(
    const float* __restrict__ rhoIn, float* __restrict__ rhoOut,
    const _Float16* __restrict__ velIn, _Float16* __restrict__ velOut) {
  __shared__ __align__(16) _Float16 sVin[32 * 272];  // staged vel comp + halo
  __shared__ __align__(16) _Float16 sBx[32 * 256];   // blur-x result
  __shared__ __align__(16) unsigned int sVn32[4096]; // new vel, 2 comps, f16x2

  const int img = blockIdx.y;
  const int rowBase = blockIdx.x * STRIP_R;
  const int tid = threadIdx.x;
  const int lane = tid & 31;
  // Wave index as a wave-uniform (SGPR) value: guarantees the WMMA tile loops
  // below compile to scalar branches, so EXEC is all-ones at every v_wmma
  // (ISA 7.12 requirement), instead of exec-masked lane loops.
  const int wave = __builtin_amdgcn_readfirstlane(tid >> 5);
  const size_t ibase = (size_t)img * 65536u;
  _Float16* sVn = (_Float16*)sVn32;

  // Prefetch density strip (gate + advection source): global_prefetch_b8.
  __builtin_prefetch(rhoIn + ibase + (size_t)rowBase * IMG_N, 0, 1);

  // Normalized Gaussian taps, sigma = 1, radius 3.
  const float e1 = expf(-0.5f), e2 = expf(-2.0f), e3 = expf(-4.5f);
  const float norm = 1.0f / (1.0f + 2.0f * (e1 + e2 + e3));
  float wk[4] = {norm, e1 * norm, e2 * norm, e3 * norm};
  auto kw = [&](int t) -> float {
    int a = t < 0 ? -t : t;
    return a <= 3 ? wk[a] : 0.0f;
  };

  // ---- constant WMMA operands (banded Toeplitz Gaussian) ----
  // Bx (blur-x B, 32x16): B[k][n] = kw(k-8-n);  lane layout: k=(lane<16?0:8)+(e<8?e:8+e)
  // Ay (blur-y A, 16x32): A[m][k] = kw(k-8-m);  lane layout: m=lane&15,
  //   k = (lane<16?0:8) + (e<8 ? e : 8+e)
  v16h Bx, Ay;
  {
    const int n = lane & 15;
    const int kbB = (lane < 16) ? 0 : 16;
    const int kbA = (lane < 16) ? 0 : 8;
#pragma unroll
    for (int e = 0; e < 16; ++e) {
      Bx[e] = (_Float16)kw((kbB >> 1) + (e < 8 ? e : 8 + e) - 8 - n);
      const int kA = kbA + (e < 8 ? e : 8 + e);
      Ay[e] = (_Float16)kw(kA - 8 - n);  // m == n == lane&15
    }
  }

  for (int comp = 0; comp < 2; ++comp) {
    const _Float16* vplane =
        velIn + (size_t)img * 2u * 65536u + (size_t)comp * 65536u;

#if FP_ASYNC_STAGE
    // ---- async staging: rows [rowBase-8, rowBase+24), 16 B per lane ----
    // Zero the +-8-column aprons (conv 'SAME' zero padding).
    for (int p = tid; p < 32 * 16; p += 256) {
      const int row = p >> 4, c = p & 15;
      const int col = (c < 8) ? c : (256 + c);  // LDS cols 0..7 / 264..271
      sVin[row * 272 + col] = (_Float16)0.0f;
    }
#pragma unroll
    for (int it = 0; it < 4; ++it) {
      const int p = tid + it * 256;     // 1024 x 16-byte chunks
      const int row = p >> 5;           // 32 chunks of 16 B per 512 B row
      const int hoff = (p & 31) * 8;    // half-element offset within row
      const int gr = rowBase - 8 + row;
      if (gr >= 0 && gr < IMG_N) {
        __builtin_amdgcn_global_load_async_to_lds_b128(
            (fp_gp128)(vplane + gr * IMG_N + hoff),
            (fp_lp128)(&sVin[row * 272 + 8 + hoff]), 0, 0);
      } else {
#pragma unroll
        for (int q = 0; q < 8; ++q)
          sVin[row * 272 + 8 + hoff + q] = (_Float16)0.0f;
      }
    }
    __builtin_amdgcn_s_wait_asynccnt(0);
#else
    // ---- scalar staging fallback ----
    for (int p = tid; p < 32 * 272; p += 256) {
      const int row = p / 272, col = p % 272;
      const int gr = rowBase - 8 + row, gc = col - 8;
      _Float16 v = (_Float16)0.0f;
      if (gr >= 0 && gr < IMG_N && gc >= 0 && gc < IMG_N)
        v = vplane[gr * IMG_N + gc];
      sVin[p] = v;
    }
#endif
    __syncthreads();

    // ---- blur-x: 2 row-tiles x 16 col-tiles over the 32-row region ----
    for (int tile = wave; tile < 32; tile += 8) {
      const int tr = tile >> 4;
      const int Jc = tile & 15;
      const int m = lane & 15;
      const int kbA = (lane < 16) ? 0 : 8;
      v16h A;
#pragma unroll
      for (int e = 0; e < 16; ++e) {
        const int k = kbA + (e < 8 ? e : 8 + e);
        A[e] = sVin[(tr * 16 + m) * 272 + Jc * 16 + k];
      }
      v8f c = {};
      c = __builtin_amdgcn_wmma_f32_16x16x32_f16(false, A, false, Bx,
                                                 (short)0, c, false, false);
      const int off8 = (lane < 16) ? 0 : 8;
#pragma unroll
      for (int v = 0; v < 8; ++v)
        sBx[(tr * 16 + v + off8) * IMG_N + Jc * 16 + m] = (_Float16)c[v];
    }
    __syncthreads();

    // ---- blur-y + decay + media gate ----
    for (int Jc = wave; Jc < 16; Jc += 8) {
      const int n = lane & 15;
      const int kbB = (lane < 16) ? 0 : 16;
      v16h Bv;
#pragma unroll
      for (int e = 0; e < 16; ++e)
        Bv[e] = sBx[((kbB >> 1) + (e < 8 ? e : 8 + e)) * IMG_N + Jc * 16 + n];
      v8f c = {};
      c = __builtin_amdgcn_wmma_f32_16x16x32_f16(false, Ay, false, Bv,
                                                 (short)0, c, false, false);
      const int off8 = (lane < 16) ? 0 : 8;
#pragma unroll
      for (int v = 0; v < 8; ++v) {
        const int m = v + off8;
        const int col = Jc * 16 + n;
        const float rc = rhoIn[ibase + (size_t)(rowBase + m) * IMG_N + col];
        const float gate = sigmoidf_(MEDIA_SHARP_C * (rc - 0.5f));
        const float vn = c[v] * DECAY_C * gate;
        sVn[comp * 4096 + m * IMG_N + col] = (_Float16)vn;
      }
    }
    __syncthreads();
  }

  // ---- coalesced velocity writeback (f16x2 words) ----
  {
    unsigned int* vout32 = (unsigned int*)velOut;
    for (int wi = tid; wi < 4096; wi += 256) {
      const int comp = wi >> 11;
      const int rem = wi & 2047;
      const int m = rem >> 7;
      const int jw = rem & 127;
      vout32[((size_t)img * 2u + comp) * 32768u +
             (size_t)(rowBase + m) * 128u + jw] = sVn32[wi];
    }
  }

  // ---- semi-Lagrangian advection with clamped bilinear gather ----
  for (int p = tid; p < STRIP_R * IMG_N; p += 256) {
    const int m = p >> 8, j = p & 255;
    const float vx = (float)sVn[m * IMG_N + j];         // along columns (X)
    const float vy = (float)sVn[4096 + m * IMG_N + j];  // along rows (Y)
    const float cb = (float)j - vx;
    const float rb = (float)(rowBase + m) - vy;
    const float cf = floorf(cb), rf = floorf(rb);
    const float wc = cb - cf, wr = rb - rf;
    const int c0i = clampi_((int)cf, 0, IMG_N - 1);
    const int c1i = clampi_((int)cf + 1, 0, IMG_N - 1);
    const int r0i = clampi_((int)rf, 0, IMG_N - 1);
    const int r1i = clampi_((int)rf + 1, 0, IMG_N - 1);
    const float* R = rhoIn + ibase;
    const float v00 = R[r0i * IMG_N + c0i];
    const float v01 = R[r0i * IMG_N + c1i];
    const float v10 = R[r1i * IMG_N + c0i];
    const float v11 = R[r1i * IMG_N + c1i];
    const float outv = v00 * (1.0f - wc) * (1.0f - wr) +
                       v01 * wc * (1.0f - wr) +
                       v10 * (1.0f - wc) * wr + v11 * wc * wr;
    rhoOut[ibase + (size_t)(rowBase + m) * IMG_N + j] = outv;
  }
}

// ---------------------------------------------------------------------------
extern "C" void kernel_launch(void* const* d_in, const int* in_sizes, int n_in,
                              void* d_out, int out_size, void* d_ws,
                              size_t ws_size, hipStream_t stream) {
  const float* occ = (const float*)d_in[0];        // (64,256,256) f32
  const float* astart = (const float*)d_in[1];     // (64,3) f32
  const float* aend = (const float*)d_in[2];       // (64,3) f32
  float* out = (float*)d_out;                      // (64,256,256) f32

  char* ws = (char*)d_ws;
  _Float16* V0 = (_Float16*)ws;                                   // 16 MiB
  _Float16* V1 = (_Float16*)(ws + (size_t)16 * 1024 * 1024);      // 16 MiB
  float* R0 = (float*)(ws + (size_t)32 * 1024 * 1024);            // 16 MiB

  fp_init_vel<<<dim3(256, 64), 256, 0, stream>>>(astart, aend, V0);

  for (int k = 0; k < N_STEPS_C; ++k) {
    // rho ping-pong chosen so step 19 (odd) lands in d_out.
    const float* rin = (k == 0) ? occ : ((k & 1) ? R0 : out);
    float* rout = (k & 1) ? out : R0;
    const _Float16* vin = (k & 1) ? V1 : V0;
    _Float16* vout = (k & 1) ? V0 : V1;
    fp_step<<<dim3(IMG_N / STRIP_R, 64), 256, 0, stream>>>(rin, rout, vin,
                                                           vout);
  }
}
